// GFlowCayleyLinear_27427661152585
// MI455X (gfx1250) — matrix operand order repaired and checked by reading, not compiled, not run on hardware
//
#include <hip/hip_runtime.h>
#include <hip/hip_bf16.h>

typedef __attribute__((ext_vector_type(16))) _Float16 v16h;
typedef __attribute__((ext_vector_type(8)))  float    v8f;

#define NA 8          // actions
#define ND 64         // embedding dim (K of layer 1)
#define NH 128        // hidden dim
#define W1T_PITCH 72  // padded LDS pitch (halves) -> conflict-free ds_load_b128

__device__ __forceinline__ float softplusf(float x) {
    // numerically stable softplus with fast transcendentals
    return fmaxf(x, 0.0f) + __logf(1.0f + __expf(-fabsf(x)));
}

__device__ __forceinline__ void stage_weights(const float* __restrict__ W1,
                                              const float* __restrict__ b1,
                                              const float* __restrict__ W2,
                                              _Float16* sW1T, float* sB1, float* sW2) {
    int t = threadIdx.x;
    // W1 [64][128] fp32 -> LDS W1T [128][72] fp16 (transposed, padded)
    for (int idx = t; idx < ND * NH; idx += (int)blockDim.x) {
        int n = idx >> 6;      // hidden index 0..127
        int k = idx & 63;      // input feature 0..63
        sW1T[n * W1T_PITCH + k] = (_Float16)W1[k * NH + n];
    }
    for (int idx = t; idx < NH * NA; idx += (int)blockDim.x) sW2[idx] = W2[idx];
    for (int idx = t; idx < NH;      idx += (int)blockDim.x) sB1[idx] = b1[idx];
    __syncthreads();
}

// A-fragment (16x32 f16): lane L holds row M=L&15; element e holds
// K = c*32 + (e>>3)*16 + (L>>4)*8 + (e&7)  ->  two contiguous 8-float runs.
__device__ __forceinline__ v16h load_a_frag(const float* __restrict__ rowp, int c, int hl) {
    const float* p0 = rowp + c * 32 + hl * 8;        // e = 0..7
    const float* p1 = rowp + c * 32 + 16 + hl * 8;   // e = 8..15
    float4 x0 = *(const float4*)(p0);
    float4 x1 = *(const float4*)(p0 + 4);
    float4 x2 = *(const float4*)(p1);
    float4 x3 = *(const float4*)(p1 + 4);
    v16h a;
    a[0]  = (_Float16)x0.x; a[1]  = (_Float16)x0.y; a[2]  = (_Float16)x0.z; a[3]  = (_Float16)x0.w;
    a[4]  = (_Float16)x1.x; a[5]  = (_Float16)x1.y; a[6]  = (_Float16)x1.z; a[7]  = (_Float16)x1.w;
    a[8]  = (_Float16)x2.x; a[9]  = (_Float16)x2.y; a[10] = (_Float16)x2.z; a[11] = (_Float16)x2.w;
    a[12] = (_Float16)x3.x; a[13] = (_Float16)x3.y; a[14] = (_Float16)x3.z; a[15] = (_Float16)x3.w;
    return a;
}

// B-fragment (32x16 f16) of W1 tile: lane L holds column N = jt*16 + (L&15);
// elements e hold K = c*32 + (e>>3)*16 + (L>>4)*8 + (e&7)  -> two 8-half runs in W1T.
__device__ __forceinline__ v16h load_b_frag_w1(const _Float16* sW1T, int jt, int c, int l15, int hl) {
    const _Float16* wp = sW1T + (jt * 16 + l15) * W1T_PITCH + c * 32 + hl * 8;
    union { v16h h; uint4 u[2]; } tmp;
    tmp.u[0] = *(const uint4*)(wp);
    tmp.u[1] = *(const uint4*)(wp + 16);
    return tmp.h;
}

// Layer 1 for a 16-row tile: acc[jt] = X(16x64) * W1[:, jt*16:+16] in fp32.
__device__ __forceinline__ void layer1(const float* __restrict__ rowp, const _Float16* sW1T,
                                       const float* b1n, int l15, int hl, v8f acc[8]) {
    #pragma unroll
    for (int jt = 0; jt < 8; ++jt) acc[jt] = (v8f){0,0,0,0,0,0,0,0};
    #pragma unroll
    for (int c = 0; c < 2; ++c) {
        v16h af = load_a_frag(rowp, c, hl);
        #pragma unroll
        for (int jt = 0; jt < 8; ++jt) {
            v16h bf = load_b_frag_w1(sW1T, jt, c, l15, hl);
            acc[jt] = __builtin_amdgcn_wmma_f32_16x16x32_f16(
                false, af, false, bf, (short)0, acc[jt], false, false);
        }
    }
    // bias + ReLU (C-layout: lane column N = jt*16+l15, VGPR r -> row r + 8*hl)
    #pragma unroll
    for (int jt = 0; jt < 8; ++jt) {
        #pragma unroll
        for (int r = 0; r < 8; ++r)
            acc[jt][r] = fmaxf(acc[jt][r] + b1n[jt], 0.0f);
    }
}

// ---------------- Fin: backward edges, diagonal output only ----------------
__global__ __launch_bounds__(256) void gflow_fin_kernel(
    const float* __restrict__ backward, const float* __restrict__ initial_flow,
    const float* __restrict__ W1, const float* __restrict__ b1,
    const float* __restrict__ W2, const float* __restrict__ b2,
    float* __restrict__ out, int tilesPerWave, int nTiles)
{
    __shared__ _Float16 sW1T[NH * W1T_PITCH];
    __shared__ float sW2[NH * NA];
    __shared__ float sB1[NH];
    stage_weights(W1, b1, W2, sW1T, sB1, sW2);

    int lane = threadIdx.x & 31, wave = threadIdx.x >> 5;
    int l15 = lane & 15, hl = lane >> 4;
    int gwave = blockIdx.x * 8 + wave;

    float b1n[8];
    #pragma unroll
    for (int jt = 0; jt < 8; ++jt) b1n[jt] = sB1[jt * 16 + l15];
    float fin_base = initial_flow[0];

    for (int it = 0; it < tilesPerWave; ++it) {
        int tile = gwave * tilesPerWave + it;           // wave-uniform guard
        if (tile >= nTiles) break;
        long long tileBase = (long long)tile * 16;      // backward row index
        long long rg = tileBase + l15;                  // lane's A row
        long long bt = rg >> 3;  int act = (int)(rg & 7);
        const float* rowp = backward + (bt * 9 + 1 + act) * (long long)ND;

        v8f acc[8];
        layer1(rowp, sW1T, b1n, l15, hl, acc);

        // layer 2, diagonal: row (tileBase + r + 8*hl) needs action a = r
        float part[8];
        #pragma unroll
        for (int r = 0; r < 8; ++r) part[r] = 0.0f;
        #pragma unroll
        for (int jt = 0; jt < 8; ++jt) {
            const float* w = sW2 + (jt * 16 + l15) * NA;
            float4 wa = *(const float4*)(w);
            float4 wb = *(const float4*)(w + 4);
            part[0] += acc[jt][0] * wa.x;  part[1] += acc[jt][1] * wa.y;
            part[2] += acc[jt][2] * wa.z;  part[3] += acc[jt][3] * wa.w;
            part[4] += acc[jt][4] * wb.x;  part[5] += acc[jt][5] * wb.y;
            part[6] += acc[jt][6] * wb.z;  part[7] += acc[jt][7] * wb.w;
        }
        // 16 lanes of each half cover all 128 hidden columns -> xor-reduce
        float s = 0.0f;
        #pragma unroll
        for (int r = 0; r < 8; ++r) {
            float v = part[r];
            v += __shfl_xor(v, 1, 32);
            v += __shfl_xor(v, 2, 32);
            v += __shfl_xor(v, 4, 32);
            v += __shfl_xor(v, 8, 32);
            s += softplusf(v + b2[r]);
        }
        if (l15 == 0) {
            long long btOut = (tileBase >> 3) + hl;     // lanes0-15 -> bt0, 16-31 -> bt1
            out[btOut * 2 + 0] = fin_base + s;
        }
    }
}

// ---------------- Fout: forward edges (slot 0), all 8 outputs ----------------
__global__ __launch_bounds__(256) void gflow_fout_kernel(
    const float* __restrict__ forward, const float* __restrict__ reward,
    const float* __restrict__ W1, const float* __restrict__ b1,
    const float* __restrict__ W2, const float* __restrict__ b2,
    float* __restrict__ out, int tilesPerWave, int nTiles)
{
    __shared__ _Float16 sW1T[NH * W1T_PITCH];
    __shared__ float sW2[NH * NA];
    __shared__ float sB1[NH];
    stage_weights(W1, b1, W2, sW1T, sB1, sW2);

    int lane = threadIdx.x & 31, wave = threadIdx.x >> 5;
    int l15 = lane & 15, hl = lane >> 4;
    int gwave = blockIdx.x * 8 + wave;

    float b1n[8];
    #pragma unroll
    for (int jt = 0; jt < 8; ++jt) b1n[jt] = sB1[jt * 16 + l15];

    for (int it = 0; it < tilesPerWave; ++it) {
        int tile = gwave * tilesPerWave + it;
        if (tile >= nTiles) break;
        long long tileBase = (long long)tile * 16;      // bt row index
        long long rg = tileBase + l15;
        const float* rowp = forward + rg * 9 * (long long)ND;  // edge slot 0

        v8f acc[8];
        layer1(rowp, sW1T, b1n, l15, hl, acc);

        float fout[8];
        #pragma unroll
        for (int r = 0; r < 8; ++r) fout[r] = 0.0f;
        #pragma unroll
        for (int a = 0; a < 8; ++a) {
            float part[8] = {0,0,0,0,0,0,0,0};
            #pragma unroll
            for (int jt = 0; jt < 8; ++jt) {
                float w = sW2[(jt * 16 + l15) * NA + a];
                #pragma unroll
                for (int r = 0; r < 8; ++r) part[r] += acc[jt][r] * w;
            }
            #pragma unroll
            for (int r = 0; r < 8; ++r) {
                float v = part[r];
                v += __shfl_xor(v, 1, 32);
                v += __shfl_xor(v, 2, 32);
                v += __shfl_xor(v, 4, 32);
                v += __shfl_xor(v, 8, 32);
                fout[r] += softplusf(v + b2[a]);
            }
        }
        if (l15 == 0) {
            #pragma unroll
            for (int r = 0; r < 8; ++r) {
                long long row = tileBase + hl * 8 + r;
                out[row * 2 + 1] = fout[r] + reward[row];
            }
        }
    }
}

extern "C" void kernel_launch(void* const* d_in, const int* in_sizes, int n_in,
                              void* d_out, int out_size, void* d_ws, size_t ws_size,
                              hipStream_t stream) {
    const float* fwd    = (const float*)d_in[0];  // [B,T,9,64]
    const float* bwd    = (const float*)d_in[1];  // [B,T,9,64]
    const float* reward = (const float*)d_in[2];  // [B,T]
    const float* iflow  = (const float*)d_in[3];  // [1]
    const float* W1     = (const float*)d_in[4];  // [64,128]
    const float* b1     = (const float*)d_in[5];  // [128]
    const float* W2     = (const float*)d_in[6];  // [128,8]
    const float* b2     = (const float*)d_in[7];  // [8]
    float* out = (float*)d_out;                   // [B,T,2]

    int BT = in_sizes[2];                         // 262144

    const int TPW = 4;                            // tiles per wave
    int tilesG  = (BT * NA) / 16;                 // 131072 backward 16-row tiles
    int blocksG = (tilesG + 8 * TPW - 1) / (8 * TPW);
    int tilesF  = BT / 16;                        // 16384 forward tiles
    int blocksF = (tilesF + 8 * TPW - 1) / (8 * TPW);

    gflow_fin_kernel<<<dim3(blocksG), dim3(256), 0, stream>>>(
        bwd, iflow, W1, b1, W2, b2, out, TPW, tilesG);
    gflow_fout_kernel<<<dim3(blocksF), dim3(256), 0, stream>>>(
        fwd, reward, W1, b1, W2, b2, out, TPW, tilesF);
}
